// MPGNO_78486232367372
// MI455X (gfx1250) — matrix operand order repaired and checked by reading, not compiled, not run on hardware
//
#include <hip/hip_runtime.h>

// ---------------------------------------------------------------------------
// MPGNO forward for MI455X (gfx1250, wave32, WMMA).
// All dense layers run through one bf16 WMMA GEMM kernel with the graph
// gather/concat fused into the A-operand load path. Weights are pre-converted
// once per call to bf16, transposed [n][k], K/N padded, so B fragments are
// direct global b128 loads (cache-resident) and only the A tile uses LDS.
// ---------------------------------------------------------------------------

typedef __attribute__((ext_vector_type(16))) __bf16 v16bf;
typedef __attribute__((ext_vector_type(8)))  float  v8f;

union ABfrag { v16bf v; unsigned int u[8]; };
union CDfrag { v8f   v; float        f[8]; };

__device__ __forceinline__ unsigned short f2bf(float f) {
  unsigned int u = __float_as_uint(f);
  unsigned int r = u + 0x7fffu + ((u >> 16) & 1u);   // round-to-nearest-even
  return (unsigned short)(r >> 16);
}

// X operand descriptor: up to 3 concatenated segments, each optionally
// gathered through a per-edge index, each with its own batch stride.
// value(row, col): b = row / rowsPerB; e = row % rowsPerB;
//   seg s with col in [off_s, off_s+l_s): r = idx_s ? idx_s[e] : e
//   val = p_s[(s_s*b + r)*l_s + (col-off_s)]
struct XDesc {
  const float* p0; const float* p1; const float* p2;
  const int*   i0; const int*   i1; const int*   i2;
  int  l0, l1, l2;
  long s0, s1, s2;
  int  rowsPerB;
};

// ---------------------------------------------------------------------------
// WMMA GEMM: Y[M,N] = act(X[M,K] @ W[K,N] + bias)
// Block: 256 threads = 8 waves. Tile: 128x128, K step 32 (bf16 WMMA).
// Wave (wm in 0..3, wn in 0..1) owns rows wm*32..+31, cols wn*64..+63
// as a 2x4 grid of 16x16 WMMA tiles. Wb is bf16 [Npad][Kpad] (transposed).
// ---------------------------------------------------------------------------
__global__ __launch_bounds__(256)
void wmma_gemm_kernel(XDesc xd, const unsigned short* __restrict__ Wb, int Kpad,
                      const float* __restrict__ bias,
                      float* __restrict__ Y,
                      int Mrows, int K, int N, int act)
{
  __shared__ unsigned short lx[2][128][40];  // double-buffered A tile [m][k] bf16

  const int tid   = threadIdx.x;
  const int lane  = tid & 31;
  const int wid   = tid >> 5;
  const int wm    = wid & 3;
  const int wn    = wid >> 2;
  const int half  = lane >> 4;
  const int l16   = lane & 15;
  const int mBlk  = blockIdx.x * 128;
  const int nBlk  = blockIdx.y * 128;

  // per-thread loop-invariant staging coords: k-column fixed, rows step by 8
  const int xc  = tid & 31;     // k within 32-wide tile
  const int xr0 = tid >> 5;     // row base (rows xr0 + 8*it, it=0..15)

  // per-wave B fragment base pointers (columns fixed across k loop)
  const unsigned short* wcol[4];
#pragma unroll
  for (int tn = 0; tn < 4; ++tn)
    wcol[tn] = Wb + (size_t)(nBlk + wn * 64 + tn * 16 + l16) * Kpad + half * 8;

  CDfrag acc[2][4];
#pragma unroll
  for (int i = 0; i < 2; ++i)
#pragma unroll
    for (int j = 0; j < 4; ++j)
#pragma unroll
      for (int r = 0; r < 8; ++r) acc[i][j].f[r] = 0.0f;

  const int kTiles = (K + 31) >> 5;
  for (int kt = 0; kt < kTiles; ++kt) {
    const int k0 = kt << 5;
    const int p  = kt & 1;

    // ---- segment select, hoisted: per thread, once per k-tile ----
    const int gc = k0 + xc;
    const float* src = xd.p0; const int* sidx = xd.i0;
    int sl = xd.l0; long ss = xd.s0; int cc = gc;
    if (gc >= xd.l0 + xd.l1) {
      src = xd.p2; sidx = xd.i2; sl = xd.l2; ss = xd.s2; cc = gc - xd.l0 - xd.l1;
    } else if (gc >= xd.l0) {
      src = xd.p1; sidx = xd.i1; sl = xd.l1; ss = xd.s1; cc = gc - xd.l0;
    }
    const bool cok = (gc < K);

    // ---- stage X tile (gather / concat fused), fp32 -> bf16 ----
    int gr = mBlk + xr0;
    int b  = gr / xd.rowsPerB;
    int e  = gr - b * xd.rowsPerB;
#pragma unroll
    for (int it = 0; it < 16; ++it) {
      float v = 0.0f;
      if (cok && gr < Mrows) {
        long row = sidx ? (long)sidx[e] : (long)e;
        v = src[(ss * (long)b + row) * (long)sl + cc];
      }
      lx[p][xr0 + (it << 3)][xc] = f2bf(v);
      gr += 8; e += 8;
      if (e >= xd.rowsPerB) { e -= xd.rowsPerB; ++b; }
    }
    __syncthreads();

    // ---- B fragments: direct global b128 pairs (bf16, transposed) ----
    ABfrag bf[4];
#pragma unroll
    for (int tn = 0; tn < 4; ++tn) {
      const uint4* q = (const uint4*)(wcol[tn] + k0);
      uint4 x0 = q[0], x1 = q[2];
      bf[tn].u[0] = x0.x; bf[tn].u[1] = x0.y; bf[tn].u[2] = x0.z; bf[tn].u[3] = x0.w;
      bf[tn].u[4] = x1.x; bf[tn].u[5] = x1.y; bf[tn].u[6] = x1.z; bf[tn].u[7] = x1.w;
    }
    // prefetch next W k-tile (lowers to global_prefetch_b8)
    if (kt + 1 < kTiles) __builtin_prefetch(wcol[0] + k0 + 32, 0, 0);

    // ---- A fragments from LDS: two b128 per 16x16 tile ----
    ABfrag a[2];
#pragma unroll
    for (int tm = 0; tm < 2; ++tm) {
      int row = wm * 32 + tm * 16 + l16;                          // M = lane%16
      const uint4* q0 = (const uint4*)&lx[p][row][half * 8];      // K grp0
      const uint4* q1 = (const uint4*)&lx[p][row][16 + half * 8]; // K grp1
      uint4 x0 = *q0, x1 = *q1;
      a[tm].u[0] = x0.x; a[tm].u[1] = x0.y; a[tm].u[2] = x0.z; a[tm].u[3] = x0.w;
      a[tm].u[4] = x1.x; a[tm].u[5] = x1.y; a[tm].u[6] = x1.z; a[tm].u[7] = x1.w;
    }

#pragma unroll
    for (int tm = 0; tm < 2; ++tm)
#pragma unroll
      for (int tn = 0; tn < 4; ++tn)
        acc[tm][tn].v = __builtin_amdgcn_wmma_f32_16x16x32_bf16(
            false, a[tm].v, false, bf[tn].v, (short)0, acc[tm][tn].v, false, false);
    // single barrier per iteration: next iteration writes the other buffer
  }

  // ---- epilogue: bias + optional swish, fp32 store ----
#pragma unroll
  for (int tm = 0; tm < 2; ++tm) {
#pragma unroll
    for (int tn = 0; tn < 4; ++tn) {
      int col = nBlk + wn * 64 + tn * 16 + l16;
      if (col >= N) continue;
      float bb = bias ? bias[col] : 0.0f;
#pragma unroll
      for (int r = 0; r < 8; ++r) {
        int row = mBlk + wm * 32 + tm * 16 + r + half * 8;   // C/D layout
        if (row < Mrows) {
          float v = acc[tm][tn].f[r] + bb;
          if (act) v = v * (1.0f / (1.0f + __expf(-v)));      // swish
          Y[(long)row * N + col] = v;
        }
      }
    }
  }
}

// ---------------------------------------------------------------------------
// Weight pre-conversion: Wb[n][k] = bf16(W[k][n]), zero-padded to Kpad/Npad.
// ---------------------------------------------------------------------------
__global__ void convw_kernel(const float* __restrict__ W, unsigned short* __restrict__ Wb,
                             int K, int N, int Kpad, long total) {
  long i = (long)blockIdx.x * blockDim.x + threadIdx.x;
  long st = (long)gridDim.x * blockDim.x;
  for (; i < total; i += st) {
    int k = (int)(i % Kpad);
    int n = (int)(i / Kpad);
    Wb[i] = (k < K && n < N) ? f2bf(W[(long)k * N + n]) : (unsigned short)0;
  }
}

// ---------------------------------------------------------------------------
// Support kernels
// ---------------------------------------------------------------------------
__global__ __launch_bounds__(128)
void ln_kernel(float* __restrict__ x, const float* __restrict__ g,
               const float* __restrict__ b, int D)
{
  __shared__ float s1[128], s2[128];
  long row = blockIdx.x;
  float* xr = x + row * (long)D;
  float sum = 0.f, sq = 0.f;
  for (int i = threadIdx.x; i < D; i += 128) { float v = xr[i]; sum += v; sq += v * v; }
  s1[threadIdx.x] = sum; s2[threadIdx.x] = sq;
  __syncthreads();
  for (int off = 64; off > 0; off >>= 1) {
    if ((int)threadIdx.x < off) {
      s1[threadIdx.x] += s1[threadIdx.x + off];
      s2[threadIdx.x] += s2[threadIdx.x + off];
    }
    __syncthreads();
  }
  float mean = s1[0] / D;
  float var  = s2[0] / D - mean * mean;
  float rstd = rsqrtf(var + 1e-6f);
  for (int i = threadIdx.x; i < D; i += 128)
    xr[i] = (xr[i] - mean) * rstd * g[i] + b[i];
}

__global__ void zero_kernel(float* p, long n) {
  long i = (long)blockIdx.x * blockDim.x + threadIdx.x;
  long st = (long)gridDim.x * blockDim.x;
  for (; i < n; i += st) p[i] = 0.0f;
}
__global__ void add_kernel(float* d, const float* s, long n) {
  long i = (long)blockIdx.x * blockDim.x + threadIdx.x;
  long st = (long)gridDim.x * blockDim.x;
  for (; i < n; i += st) d[i] += s[i];
}
__global__ void copy_kernel(float* d, const float* s, long n) {
  long i = (long)blockIdx.x * blockDim.x + threadIdx.x;
  long st = (long)gridDim.x * blockDim.x;
  for (; i < n; i += st) d[i] = s[i];
}
__global__ void concat2_kernel(float* out, const float* a, const float* b,
                               int Dw, long total) {  // out width = 2*Dw
  long i = (long)blockIdx.x * blockDim.x + threadIdx.x;
  long st = (long)gridDim.x * blockDim.x;
  int W2 = 2 * Dw;
  for (; i < total; i += st) {
    int  c = (int)(i % W2);
    long r = i / W2;
    out[i] = (c < Dw) ? a[r * Dw + c] : b[r * Dw + (c - Dw)];
  }
}
__global__ void count_kernel(const int* idx, float* cnt, int E) {
  int i = blockIdx.x * blockDim.x + threadIdx.x;
  if (i < E) atomicAdd(&cnt[idx[i]], 1.0f);
}
__global__ void scatter_add_kernel(const float* data, const int* idx, float* out,
                                   int E, int Dw, int Nseg, long total) {
  long i = (long)blockIdx.x * blockDim.x + threadIdx.x;
  long st = (long)gridDim.x * blockDim.x;
  for (; i < total; i += st) {
    int  d  = (int)(i % Dw);
    long be = i / Dw;
    int  e  = (int)(be % E);
    int  b  = (int)(be / E);
    atomicAdd(&out[((long)b * Nseg + idx[e]) * Dw + d], data[i]);
  }
}
__global__ void div_kernel(float* out, const float* cnt, int Nseg, int Dw, long total) {
  long i = (long)blockIdx.x * blockDim.x + threadIdx.x;
  long st = (long)gridDim.x * blockDim.x;
  for (; i < total; i += st) {
    long bn = i / Dw;
    int  n  = (int)(bn % Nseg);
    out[i] /= fmaxf(cnt[n], 1.0f);
  }
}

// ---------------------------------------------------------------------------
// Host orchestration
// ---------------------------------------------------------------------------
struct BlockP {
  const float* g; const float* be;
  const float* W0; const float* b0;
  const float* W1; const float* b1;
  const float* W2; const float* b2;
  int din, dout;
};
struct GW { const unsigned short* w; int Kpad; };
struct BlockG {
  GW W0, W1, W2;
  const float* b0; const float* b1; const float* b2;
  const float* g; const float* be;
  int din, dout;
};

static BlockP take_block(const float*& c, int din, int dout) {
  BlockP bk; bk.din = din; bk.dout = dout;
  bk.g  = c; c += dout;
  bk.be = c; c += dout;
  bk.W0 = c; c += (long)din * 128;  bk.b0 = c; c += 128;
  bk.W1 = c; c += 128 * 128;        bk.b1 = c; c += 128;
  bk.W2 = c; c += (long)128 * dout; bk.b2 = c; c += dout;
  return bk;
}

static inline unsigned eb(long n) {
  long b = (n + 255) / 256;
  if (b > 32768) b = 32768;
  if (b < 1) b = 1;
  return (unsigned)b;
}

extern "C" void kernel_launch(void* const* d_in, const int* in_sizes, int n_in,
                              void* d_out, int out_size, void* d_ws, size_t ws_size,
                              hipStream_t stream) {
  (void)n_in; (void)out_size; (void)ws_size;
  const float* u        = (const float*)d_in[0];
  const float* grid_nf  = (const float*)d_in[1];
  const float* mesh_nf  = (const float*)d_in[2];
  const float* g2m_ef   = (const float*)d_in[3];
  const float* mesh_ef  = (const float*)d_in[4];
  const float* m2g_ef   = (const float*)d_in[5];
  const float* g2g_ef   = (const float*)d_in[6];
  const float* params   = (const float*)d_in[7];
  const int* g2m_send   = (const int*)d_in[8];
  const int* g2m_recv   = (const int*)d_in[9];
  const int* mesh_send  = (const int*)d_in[10];
  const int* mesh_recv  = (const int*)d_in[11];
  const int* m2g_send   = (const int*)d_in[12];
  const int* m2g_recv   = (const int*)d_in[13];
  const int* g2g_send   = (const int*)d_in[14];
  const int* g2g_recv   = (const int*)d_in[15];

  const int Bsz = 2, G = 128 * 128, Mn = 32 * 32, L = 128;
  const int E1 = in_sizes[8];    // g2m/m2g edges
  const int Em = in_sizes[10];   // mesh edges (4096)
  const int Eg = in_sizes[14];   // g2g edges (65536)

  // ---- parse params (jax pytree flatten: sorted dict keys) ----
  const float* cur = params;
  BlockP g2g_ee_p = take_block(cur, 3, 128);
  const float* oW0 = cur; cur += 256 * 128; const float* ob0 = cur; cur += 128;
  const float* oW1 = cur; cur += 128 * 128; const float* ob1 = cur; cur += 128;
  const float* oW2 = cur; cur += 128 * 2;   const float* ob2 = cur; cur += 2;
  BlockP g2g_e_p[2], g2g_n_p[2];
  for (int s = 0; s < 2; ++s) { g2g_e_p[s] = take_block(cur, 640, 128); g2g_n_p[s] = take_block(cur, 384, 256); }
  BlockP g2m_ee_p = take_block(cur, 3, 128);
  BlockP g2m_eu_p = take_block(cur, 384, 128);
  BlockP g2m_ge_p = take_block(cur, 6, 128);
  BlockP g2m_gu_p = take_block(cur, 128, 128);
  BlockP g2m_me_p = take_block(cur, 4, 128);
  BlockP g2m_mu_p = take_block(cur, 256, 128);
  BlockP m2g_ee_p = take_block(cur, 3, 128);
  BlockP m2g_eu_p = take_block(cur, 384, 128);
  BlockP m2g_gu_p = take_block(cur, 256, 128);
  BlockP msh_ee_p = take_block(cur, 3, 128);
  BlockP msh_e_p[18], msh_n_p[18];
  for (int s = 0; s < 18; ++s) { msh_e_p[s] = take_block(cur, 384, 128); msh_n_p[s] = take_block(cur, 256, 128); }

  // ---- workspace bump allocator ----
  char* wp = (char*)d_ws;
  auto allocB = [&](long nbytes) { char* r = wp; wp += ((nbytes + 255) / 256) * 256; return r; };
  auto alloc  = [&](long nfloats) { return (float*)allocB(nfloats * 4); };

  // ---- pre-convert all weights to bf16, transposed [Npad][Kpad] ----
  auto prepW = [&](const float* W, int K, int N) {
    int Kpad = ((K + 31) / 32) * 32;
    int Npad = ((N + 127) / 128) * 128;
    unsigned short* wb = (unsigned short*)allocB((long)Npad * Kpad * 2);
    long tot = (long)Npad * Kpad;
    convw_kernel<<<eb(tot), 256, 0, stream>>>(W, wb, K, N, Kpad, tot);
    GW g; g.w = wb; g.Kpad = Kpad; return g;
  };
  auto prep_block = [&](const BlockP& p) {
    BlockG g;
    g.W0 = prepW(p.W0, p.din, 128);  g.b0 = p.b0;
    g.W1 = prepW(p.W1, 128, 128);    g.b1 = p.b1;
    g.W2 = prepW(p.W2, 128, p.dout); g.b2 = p.b2;
    g.g = p.g; g.be = p.be; g.din = p.din; g.dout = p.dout;
    return g;
  };
  BlockG g2g_ee = prep_block(g2g_ee_p);
  BlockG g2g_e[2], g2g_n[2];
  for (int s = 0; s < 2; ++s) { g2g_e[s] = prep_block(g2g_e_p[s]); g2g_n[s] = prep_block(g2g_n_p[s]); }
  BlockG g2m_ee = prep_block(g2m_ee_p), g2m_eu = prep_block(g2m_eu_p);
  BlockG g2m_ge = prep_block(g2m_ge_p), g2m_gu = prep_block(g2m_gu_p);
  BlockG g2m_me = prep_block(g2m_me_p), g2m_mu = prep_block(g2m_mu_p);
  BlockG m2g_ee = prep_block(m2g_ee_p), m2g_eu = prep_block(m2g_eu_p);
  BlockG m2g_gu = prep_block(m2g_gu_p);
  BlockG msh_ee = prep_block(msh_ee_p);
  BlockG msh_e[18], msh_n[18];
  for (int s = 0; s < 18; ++s) { msh_e[s] = prep_block(msh_e_p[s]); msh_n[s] = prep_block(msh_n_p[s]); }
  GW oG0 = prepW(oW0, 256, 128), oG1 = prepW(oW1, 128, 128), oG2 = prepW(oW2, 128, 2);

  // ---- activation buffers ----
  float* grid_lat  = alloc((long)Bsz * G * L);
  float* grid_lat0 = alloc((long)Bsz * G * L);
  float* mesh_lat  = alloc((long)Bsz * Mn * L);
  float* h         = alloc((long)Bsz * G * 256);
  float* e_lat     = alloc((long)Bsz * E1 * L);
  float* me        = alloc((long)Bsz * Em * L);
  float* gge       = alloc((long)Bsz * Eg * L);
  float* seg       = alloc((long)Bsz * G * L);
  float* cnt       = alloc(G);
  float* t1        = alloc((long)Bsz * Eg * L);
  float* t2        = alloc((long)Bsz * Eg * L);
  float* t3        = alloc((long)Bsz * Eg * L);

  auto gemm = [&](const XDesc& xd, const GW& W, const float* bias, float* Y,
                  long Mr, int K, int N, int act) {
    dim3 g((unsigned)((Mr + 127) / 128), (unsigned)((N + 127) / 128));
    wmma_gemm_kernel<<<g, 256, 0, stream>>>(xd, W.w, W.Kpad, bias, Y, (int)Mr, K, N, act);
  };
  auto plainX = [&](const float* p, int Lw, long Mr) {
    XDesc xd{}; xd.p0 = p; xd.l0 = Lw; xd.s0 = 0; xd.rowsPerB = (int)Mr;
    return xd;
  };
  auto run_block = [&](const BlockG& bk, const XDesc& xin, long Mr, float* Y) {
    gemm(xin, bk.W0, bk.b0, t1, Mr, bk.din, 128, 1);
    gemm(plainX(t1, 128, Mr), bk.W1, bk.b1, t2, Mr, 128, 128, 1);
    gemm(plainX(t2, 128, Mr), bk.W2, bk.b2, Y, Mr, 128, bk.dout, 0);
    ln_kernel<<<(unsigned)Mr, 128, 0, stream>>>(Y, bk.g, bk.be, bk.dout);
  };
  auto addi = [&](float* d, const float* s, long n) {
    add_kernel<<<eb(n), 256, 0, stream>>>(d, s, n);
  };
  auto seg_mean = [&](const float* data, const int* idx, int E, int Nseg, float* outp) {
    long totO = (long)Bsz * Nseg * L;
    zero_kernel<<<eb(totO), 256, 0, stream>>>(outp, totO);
    zero_kernel<<<eb(Nseg), 256, 0, stream>>>(cnt, Nseg);
    count_kernel<<<(E + 255) / 256, 256, 0, stream>>>(idx, cnt, E);
    long totD = (long)Bsz * E * L;
    scatter_add_kernel<<<eb(totD), 256, 0, stream>>>(data, idx, outp, E, L, Nseg, totD);
    div_kernel<<<eb(totO), 256, 0, stream>>>(outp, cnt, Nseg, L, totO);
  };

  // ================= encoder (grid2mesh) =================
  { XDesc x{}; x.p0 = u; x.l0 = 2; x.s0 = G;        // [u | grid_nfeat]
    x.p1 = grid_nf; x.l1 = 4; x.s1 = 0;
    x.rowsPerB = G;
    run_block(g2m_ge, x, (long)Bsz * G, grid_lat); }
  { XDesc x{}; x.p0 = mesh_nf; x.l0 = 4; x.s0 = 0; x.rowsPerB = Mn;
    run_block(g2m_me, x, (long)Bsz * Mn, mesh_lat); }
  { XDesc x{}; x.p0 = g2m_ef; x.l0 = 3; x.s0 = 0; x.rowsPerB = E1;
    run_block(g2m_ee, x, (long)Bsz * E1, e_lat); }
  { XDesc x{};                                        // [e | grid[send] | mesh[recv]]
    x.p0 = e_lat;    x.l0 = L; x.s0 = E1;
    x.p1 = grid_lat; x.l1 = L; x.s1 = G;  x.i1 = g2m_send;
    x.p2 = mesh_lat; x.l2 = L; x.s2 = Mn; x.i2 = g2m_recv;
    x.rowsPerB = E1;
    run_block(g2m_eu, x, (long)Bsz * E1, t3);
    addi(e_lat, t3, (long)Bsz * E1 * L); }
  seg_mean(e_lat, g2m_recv, E1, Mn, seg);
  { XDesc x{}; x.p0 = mesh_lat; x.l0 = L; x.s0 = Mn;  // [mesh | mean]
    x.p1 = seg; x.l1 = L; x.s1 = Mn; x.rowsPerB = Mn;
    run_block(g2m_mu, x, (long)Bsz * Mn, t3);
    addi(mesh_lat, t3, (long)Bsz * Mn * L); }
  { run_block(g2m_gu, plainX(grid_lat, 128, (long)Bsz * G), (long)Bsz * G, t3);
    addi(grid_lat, t3, (long)Bsz * G * L); }
  copy_kernel<<<eb((long)Bsz * G * L), 256, 0, stream>>>(grid_lat0, grid_lat, (long)Bsz * G * L);

  // ================= processor: 18 mesh steps =================
  { XDesc x{}; x.p0 = mesh_ef; x.l0 = 3; x.s0 = 0; x.rowsPerB = Em;
    run_block(msh_ee, x, (long)Bsz * Em, me); }
  for (int s = 0; s < 18; ++s) {
    XDesc xe{};
    xe.p0 = me;       xe.l0 = L; xe.s0 = Em;
    xe.p1 = mesh_lat; xe.l1 = L; xe.s1 = Mn; xe.i1 = mesh_send;
    xe.p2 = mesh_lat; xe.l2 = L; xe.s2 = Mn; xe.i2 = mesh_recv;
    xe.rowsPerB = Em;
    run_block(msh_e[s], xe, (long)Bsz * Em, t3);
    addi(me, t3, (long)Bsz * Em * L);
    seg_mean(me, mesh_recv, Em, Mn, seg);
    XDesc xn{};
    xn.p0 = mesh_lat; xn.l0 = L; xn.s0 = Mn;
    xn.p1 = seg;      xn.l1 = L; xn.s1 = Mn;
    xn.rowsPerB = Mn;
    run_block(msh_n[s], xn, (long)Bsz * Mn, t3);
    addi(mesh_lat, t3, (long)Bsz * Mn * L);
  }

  // ================= decoder (mesh2grid) =================
  { XDesc x{}; x.p0 = m2g_ef; x.l0 = 3; x.s0 = 0; x.rowsPerB = E1;
    run_block(m2g_ee, x, (long)Bsz * E1, e_lat); }
  { XDesc x{};
    x.p0 = e_lat;    x.l0 = L; x.s0 = E1;
    x.p1 = mesh_lat; x.l1 = L; x.s1 = Mn; x.i1 = m2g_send;
    x.p2 = grid_lat; x.l2 = L; x.s2 = G;  x.i2 = m2g_recv;
    x.rowsPerB = E1;
    run_block(m2g_eu, x, (long)Bsz * E1, t3);
    addi(e_lat, t3, (long)Bsz * E1 * L); }
  seg_mean(e_lat, m2g_recv, E1, G, seg);
  { XDesc x{}; x.p0 = grid_lat; x.l0 = L; x.s0 = G;
    x.p1 = seg; x.l1 = L; x.s1 = G; x.rowsPerB = G;
    run_block(m2g_gu, x, (long)Bsz * G, t3);
    addi(grid_lat, t3, (long)Bsz * G * L); }

  // ================= grid2grid refinement =================
  concat2_kernel<<<eb((long)Bsz * G * 256), 256, 0, stream>>>(
      h, grid_lat, grid_lat0, L, (long)Bsz * G * 256);
  { XDesc x{}; x.p0 = g2g_ef; x.l0 = 3; x.s0 = 0; x.rowsPerB = Eg;
    run_block(g2g_ee, x, (long)Bsz * Eg, gge); }
  for (int s = 0; s < 2; ++s) {
    XDesc xe{};                                     // [gge | h[send] | h[recv]] = 640
    xe.p0 = gge; xe.l0 = L;   xe.s0 = Eg;
    xe.p1 = h;   xe.l1 = 256; xe.s1 = G; xe.i1 = g2g_send;
    xe.p2 = h;   xe.l2 = 256; xe.s2 = G; xe.i2 = g2g_recv;
    xe.rowsPerB = Eg;
    run_block(g2g_e[s], xe, (long)Bsz * Eg, t3);
    addi(gge, t3, (long)Bsz * Eg * L);
    seg_mean(gge, g2g_recv, Eg, G, seg);
    XDesc xn{};                                     // [h | mean] = 384 -> 256
    xn.p0 = h;   xn.l0 = 256; xn.s0 = G;
    xn.p1 = seg; xn.l1 = L;   xn.s1 = G;
    xn.rowsPerB = G;
    run_block(g2g_n[s], xn, (long)Bsz * G, t3);
    addi(h, t3, (long)Bsz * G * 256);
  }

  // ================= output head (no LN) =================
  long MG = (long)Bsz * G;
  gemm(plainX(h, 256, MG), oG0, ob0, t1, MG, 256, 128, 1);
  gemm(plainX(t1, 128, MG), oG1, ob1, t2, MG, 128, 128, 1);
  gemm(plainX(t2, 128, MG), oG2, ob2, (float*)d_out, MG, 128, 2, 0);
}
